// TokenPruner_65051574665606
// MI455X (gfx1250) — compile-verified, matrix operand order checked by reading, not run on hardware
//
#include <hip/hip_runtime.h>
#include <hip/hip_bf16.h>
#include <stdint.h>

// Problem constants (from reference setup_inputs / KEEP_RATIO)
#define BB 16
#define NN 4096
#define DD 1024
#define KK 1024          // N * 0.25
#define MM 3072          // N - K
#define EPSV 1e-6f
#define NORM_EPSV 1e-12f

typedef __attribute__((ext_vector_type(16))) _Float16 v16h;
typedef __attribute__((ext_vector_type(8)))  _Float16 v8h;
typedef __attribute__((ext_vector_type(8)))  float    v8f;

// Type expected by the async-to-LDS builtin (per hipcc diagnostic):
// parameter 0: AS1 pointer to 128-bit int vector; parameter 1: AS3 pointer.
typedef int v4i_vs __attribute__((vector_size(4 * sizeof(int))));
#define AS_GLOBAL __attribute__((address_space(1)))
#define AS_LDS    __attribute__((address_space(3)))

#if __has_builtin(__builtin_amdgcn_global_load_async_to_lds_b128) && \
    __has_builtin(__builtin_amdgcn_s_wait_asynccnt)
#define USE_ASYNC_LDS 1
#else
#define USE_ASYNC_LDS 0
#endif

// ---------------------------------------------------------------------------
// 1) Top-K by rank counting. rank(i) = #{j : s_j > s_i  ||  (s_j==s_i && j<i)}
//    kept iff rank < K. Matches jax.lax.top_k tie-breaking (lower index wins).
// ---------------------------------------------------------------------------
__global__ void rank_kernel(const float* __restrict__ scores,
                            int* __restrict__ flags) {
    __shared__ float s[NN];                 // 16 KB
    const int b   = blockIdx.x >> 4;        // 16 segments of 256 per batch row
    const int seg = blockIdx.x & 15;
    const float* row = scores + b * NN;
    for (int j = threadIdx.x; j < NN; j += 256) s[j] = row[j];
    __syncthreads();
    const int i  = seg * 256 + (int)threadIdx.x;
    const float si = s[i];
    int rank = 0;
    for (int j = 0; j < NN; ++j) {
        const float sj = s[j];
        rank += (sj > si) || (sj == si && j < i);
    }
    flags[b * NN + i] = (rank < KK) ? 1 : 0;
}

// ---------------------------------------------------------------------------
// 2) Stable compaction (ascending index) of kept / dropped sets.
//    One wave per batch; wave32 ballot + popc prefix sums.
// ---------------------------------------------------------------------------
__global__ __launch_bounds__(32) void compact_kernel(
    const float* __restrict__ scores, const int* __restrict__ flags,
    int* __restrict__ kmap, int* __restrict__ dmap,
    float* __restrict__ kscore, float* __restrict__ dsc) {
    const int b    = blockIdx.x;
    const int lane = threadIdx.x;
    const unsigned lt = (1u << lane) - 1u;
    int kc = 0, dc = 0;
    for (int c = 0; c < NN / 32; ++c) {
        const int i = c * 32 + lane;
        const int f = flags[b * NN + i];
        const unsigned mask = (unsigned)__ballot(f);   // wave32: low 32 bits
        const float sc = scores[b * NN + i];
        if (f) {
            const int p = kc + __popc(mask & lt);
            kmap[b * KK + p]   = i;
            kscore[b * KK + p] = sc;                    // raw score (output)
        } else {
            const int p = dc + __popc((~mask) & lt);
            dmap[b * MM + p] = i;
            dsc[b * MM + p]  = fmaxf(sc, 0.0f);         // relu'd dropped score
        }
        const int nk = __popc(mask);
        kc += nk;
        dc += 32 - nk;
    }
}

// ---------------------------------------------------------------------------
// 3) Gather + L2-normalize -> f16 (WMMA operand precision). One wave / token.
// ---------------------------------------------------------------------------
__global__ void normalize_kernel(const float* __restrict__ vis,
                                 const int* __restrict__ map,
                                 _Float16* __restrict__ dst, int T) {
    const int wave = threadIdx.x >> 5;          // 8 waves per block
    const int lane = threadIdx.x & 31;
    const int t    = blockIdx.x * 8 + wave;     // token slot over B*T
    const int b    = t / T;
    const int slot = t % T;
    const int idx  = map[b * T + slot];
    const float* src = vis + ((size_t)b * NN + idx) * DD;
    float x[32];
    float ss = 0.0f;
    #pragma unroll
    for (int j = 0; j < 32; ++j) {
        x[j] = src[lane + 32 * j];
        ss += x[j] * x[j];
    }
    #pragma unroll
    for (int m = 16; m >= 1; m >>= 1) ss += __shfl_xor(ss, m, 32);
    const float inv = 1.0f / fmaxf(sqrtf(ss), NORM_EPSV);
    _Float16* drow = dst + ((size_t)b * T + slot) * DD;
    #pragma unroll
    for (int j = 0; j < 32; ++j) drow[lane + 32 * j] = (_Float16)(x[j] * inv);
}

// ---------------------------------------------------------------------------
// 4) sim = d_norm @ k_norm^T + argmax over K, via v_wmma_f32_16x16x32_f16.
//    One wave per 16-row dropped tile; A tile (16x1024 f16 = 32 KB) staged in
//    LDS (async-to-LDS when available) and reused across all 64 kept n-tiles.
//    Per-lane K-chunk interleave follows the ISA 16-bit A/B operand layout
//    (lanes 0-15: K 0-7 & 16-23, lanes 16-31: K 8-15 & 24-31). A WMMA B
//    column n = kept token n's contiguous d-slice, so no transpose needed.
//    Two independent accumulator chains (c0/c1) hide XDL accumulate latency.
// ---------------------------------------------------------------------------
__global__ __launch_bounds__(32) void argmax_kernel(
    const _Float16* __restrict__ dn,   // [B,M,D] dropped normalized f16
    const _Float16* __restrict__ kn,   // [B,K,D] kept normalized f16
    int* __restrict__ assign) {        // [B,M]
    __shared__ _Float16 As[16 * DD];   // 32 KB, one wave's A tile
    const int g    = blockIdx.x;            // B * M/16 = 3072 tiles
    const int b    = g / (MM / 16);
    const int mt   = g % (MM / 16);
    const int lane = threadIdx.x;
    const int half = lane >> 4;
    const int l15  = lane & 15;

    // Stage A tile row-major into LDS, 64 x 128-bit chunks per lane.
    const _Float16* abase = dn + ((size_t)b * MM + mt * 16) * DD;
#if USE_ASYNC_LDS
    for (int c = lane; c < 16 * DD / 8; c += 32)
        __builtin_amdgcn_global_load_async_to_lds_b128(
            (AS_GLOBAL v4i_vs*)(AS_GLOBAL void*)(abase + c * 8),
            (AS_LDS    v4i_vs*)(AS_LDS    void*)(As + c * 8),
            0, 0);
    __builtin_amdgcn_s_wait_asynccnt(0);
#else
    for (int c = lane; c < 16 * DD / 8; c += 32)
        *(v8h*)(As + c * 8) = *(const v8h*)(abase + c * 8);
#endif
    __syncthreads();   // single-wave WG: cheap; orders staging vs LDS reads

    float bv[8];
    int   bi[8];
    #pragma unroll
    for (int r = 0; r < 8; ++r) { bv[r] = -3.0e38f; bi[r] = 0; }

    for (int nt = 0; nt < KK / 16; ++nt) {
        // lane's B column = kept token (nt*16 + l15), chunk offset by half
        const _Float16* bbase =
            kn + ((size_t)b * KK + nt * 16 + l15) * DD + half * 8;
        if (nt + 1 < KK / 16)
            __builtin_prefetch(bbase + 16 * DD, 0, 1);   // global_prefetch_b8
        v8f c0 = {};
        v8f c1 = {};
        #pragma unroll 2
        for (int kk = 0; kk < DD / 32; kk += 2) {
            {   // even k-step -> accumulator chain 0
                const int kb = kk * 32;
                v8h a0 = *(const v8h*)(As + l15 * DD + kb + half * 8);
                v8h a1 = *(const v8h*)(As + l15 * DD + kb + 16 + half * 8);
                v16h a = __builtin_shufflevector(a0, a1, 0, 1, 2, 3, 4, 5, 6,
                                                 7, 8, 9, 10, 11, 12, 13, 14,
                                                 15);
                v8h b0 = *(const v8h*)(bbase + kb);
                v8h b1 = *(const v8h*)(bbase + kb + 16);
                v16h bm = __builtin_shufflevector(b0, b1, 0, 1, 2, 3, 4, 5, 6,
                                                  7, 8, 9, 10, 11, 12, 13, 14,
                                                  15);
                c0 = __builtin_amdgcn_wmma_f32_16x16x32_f16(
                         false, a, false, bm, (short)0, c0, false, false);
            }
            {   // odd k-step -> accumulator chain 1 (independent of chain 0)
                const int kb = kk * 32 + 32;
                v8h a0 = *(const v8h*)(As + l15 * DD + kb + half * 8);
                v8h a1 = *(const v8h*)(As + l15 * DD + kb + 16 + half * 8);
                v16h a = __builtin_shufflevector(a0, a1, 0, 1, 2, 3, 4, 5, 6,
                                                 7, 8, 9, 10, 11, 12, 13, 14,
                                                 15);
                v8h b0 = *(const v8h*)(bbase + kb);
                v8h b1 = *(const v8h*)(bbase + kb + 16);
                v16h bm = __builtin_shufflevector(b0, b1, 0, 1, 2, 3, 4, 5, 6,
                                                  7, 8, 9, 10, 11, 12, 13, 14,
                                                  15);
                c1 = __builtin_amdgcn_wmma_f32_16x16x32_f16(
                         false, a, false, bm, (short)0, c1, false, false);
            }
        }
        // C layout: VGPR r, lanes 0-15 -> row r, lanes 16-31 -> row r+8;
        // column = l15 within this tile. Strict '>' keeps the earliest max.
        const int ncol = nt * 16 + l15;
        #pragma unroll
        for (int r = 0; r < 8; ++r) {
            const float v = c0[r] + c1[r];
            if (v > bv[r]) { bv[r] = v; bi[r] = ncol; }
        }
    }

    // argmax across the 16 lanes of each half (cols mod-16 classes merge);
    // tie-break: lower kept index (jnp.argmax first-occurrence semantics).
    #pragma unroll
    for (int r = 0; r < 8; ++r) {
        #pragma unroll
        for (int mk = 8; mk >= 1; mk >>= 1) {
            const float ov = __shfl_xor(bv[r], mk, 32);
            const int   oi = __shfl_xor(bi[r], mk, 32);
            if (ov > bv[r] || (ov == bv[r] && oi < bi[r])) {
                bv[r] = ov; bi[r] = oi;
            }
        }
    }
    if (l15 == 0) {
        #pragma unroll
        for (int r = 0; r < 8; ++r)
            assign[(size_t)b * MM + mt * 16 + half * 8 + r] = bi[r];
    }
}

// ---------------------------------------------------------------------------
// 5) Deterministic merge: block per (b,k) scans assign[] (uniform scalar
//    loads) and gathers its dropped tokens in ascending m order — avoids
//    nondeterministic float atomics. Then writes all three outputs.
// ---------------------------------------------------------------------------
__global__ void merge_kernel(const float* __restrict__ vis,
                             const int* __restrict__ kmap,
                             const int* __restrict__ dmap,
                             const float* __restrict__ kscore,
                             const float* __restrict__ dsc,
                             const int* __restrict__ assign,
                             float* __restrict__ out_tok,
                             float* __restrict__ out_idx,
                             float* __restrict__ out_sc) {
    const int b   = blockIdx.x / KK;
    const int k   = blockIdx.x % KK;
    const int tid = threadIdx.x;
    float acc[4] = {0.f, 0.f, 0.f, 0.f};
    float sacc = 0.f;
    for (int m = 0; m < MM; ++m) {
        if (assign[b * MM + m] == k) {
            const float w = dsc[b * MM + m];
            const float* srow =
                vis + ((size_t)b * NN + dmap[b * MM + m]) * DD;
            #pragma unroll
            for (int j = 0; j < 4; ++j)
                acc[j] += srow[tid + 256 * j] * w;
            sacc += w;
        }
    }
    const float ks_raw = kscore[b * KK + k];
    const float ksc    = fmaxf(ks_raw, 0.0f);
    const float denom  = fmaxf(ksc + sacc, EPSV);
    const float* krow  = vis + ((size_t)b * NN + kmap[b * KK + k]) * DD;
    float* orow        = out_tok + ((size_t)b * KK + k) * DD;
    #pragma unroll
    for (int j = 0; j < 4; ++j)
        orow[tid + 256 * j] = (krow[tid + 256 * j] * ksc + acc[j]) / denom;
    if (tid == 0) {
        out_idx[b * KK + k] = (float)kmap[b * KK + k];
        out_sc[b * KK + k]  = ks_raw;
    }
}

// ---------------------------------------------------------------------------
extern "C" void kernel_launch(void* const* d_in, const int* in_sizes, int n_in,
                              void* d_out, int out_size, void* d_ws,
                              size_t ws_size, hipStream_t stream) {
    const float* vis    = (const float*)d_in[0];   // [B,N,D] f32
    const float* scores = (const float*)d_in[1];   // [B,N]   f32
    (void)in_sizes; (void)n_in; (void)out_size; (void)ws_size;

    // Workspace layout (16B-aligned chunks; ~129 MB total)
    char* ws = (char*)d_ws;
    size_t off = 0;
    _Float16* kept_norm = (_Float16*)(ws + off); off += (size_t)BB * KK * DD * 2; // 32 MB
    _Float16* drop_norm = (_Float16*)(ws + off); off += (size_t)BB * MM * DD * 2; // 96 MB
    int*   flags  = (int*)  (ws + off); off += (size_t)BB * NN * 4;
    int*   kmap   = (int*)  (ws + off); off += (size_t)BB * KK * 4;
    int*   dmap   = (int*)  (ws + off); off += (size_t)BB * MM * 4;
    float* kscore = (float*)(ws + off); off += (size_t)BB * KK * 4;
    float* dsc    = (float*)(ws + off); off += (size_t)BB * MM * 4;
    int*   assign = (int*)  (ws + off); off += (size_t)BB * MM * 4;

    float* out_tok = (float*)d_out;                       // [B,K,D]
    float* out_idx = out_tok + (size_t)BB * KK * DD;      // [B,K] (as float)
    float* out_sc  = out_idx + (size_t)BB * KK;           // [B,K]

    rank_kernel<<<dim3(BB * 16), dim3(256), 0, stream>>>(scores, flags);
    compact_kernel<<<dim3(BB), dim3(32), 0, stream>>>(scores, flags, kmap,
                                                      dmap, kscore, dsc);
    normalize_kernel<<<dim3(BB * KK / 8), dim3(256), 0, stream>>>(
        vis, kmap, kept_norm, KK);
    normalize_kernel<<<dim3(BB * MM / 8), dim3(256), 0, stream>>>(
        vis, dmap, drop_norm, MM);
    argmax_kernel<<<dim3(BB * (MM / 16)), dim3(32), 0, stream>>>(
        drop_norm, kept_norm, assign);
    merge_kernel<<<dim3(BB * KK), dim3(256), 0, stream>>>(
        vis, kmap, dmap, kscore, dsc, assign, out_tok, out_idx, out_sc);
}
